// GlobalLocalStandardAttention_84945863180765
// MI455X (gfx1250) — compile-verified, matrix operand order checked by reading, not run on hardware
//
#include <hip/hip_runtime.h>
#include <hip/hip_bf16.h>
#include <math.h>

// Problem constants (from reference)
#define BB   4
#define SS   4096
#define HH   1024
#define NHH  16
#define DD   64
#define MM   (BB*SS)          // 16384 rows
#define ATT_SCALE 0.125f      // 1/sqrt(64)

typedef __attribute__((ext_vector_type(16))) __bf16 v16bf;
typedef __attribute__((ext_vector_type(8)))  float  v8f;

#define TILE_M 128
#define TILE_N 128
#define TILE_K 32
#define LDSS   40   // LDS row stride in bf16 elems (32 data + 8 pad -> 80B rows, 16B aligned)

// ---------------------------------------------------------------------------
// Fused QKV projection: Out = X @ W^T + b, computed with v_wmma_f32_16x16x32_bf16
// grid: (HH/TILE_N, MM/TILE_M, 3)  block: 256 (8 waves of 32)
// ---------------------------------------------------------------------------
__global__ __launch_bounds__(256)
void qkv_gemm_wmma(const float* __restrict__ X,
                   const float* __restrict__ Wq, const float* __restrict__ bq,
                   const float* __restrict__ Wk, const float* __restrict__ bk,
                   const float* __restrict__ Wv, const float* __restrict__ bv,
                   float* __restrict__ Qf, float* __restrict__ Kf, float* __restrict__ Vf)
{
  __shared__ __align__(16) __bf16 As[2][TILE_M][LDSS];
  __shared__ __align__(16) __bf16 Bs[2][TILE_N][LDSS];

  const int which = blockIdx.z;
  const float* __restrict__ W    = (which == 0) ? Wq : (which == 1) ? Wk : Wv;
  const float* __restrict__ bias = (which == 0) ? bq : (which == 1) ? bk : bv;
  float* __restrict__ Out        = (which == 0) ? Qf : (which == 1) ? Kf : Vf;

  const int n0   = blockIdx.x * TILE_N;
  const int m0   = blockIdx.y * TILE_M;
  const int tid  = threadIdx.x;
  const int wave = tid >> 5;
  const int lane = tid & 31;
  const int half = lane >> 4;     // 0: lanes 0-15, 1: lanes 16-31
  const int r16  = lane & 15;
  const int waveM = wave >> 2;    // 0..1  -> 64-row slab
  const int waveN = wave & 3;     // 0..3  -> 32-col slab

  v8f acc[4][2];
  #pragma unroll
  for (int i = 0; i < 4; ++i)
    #pragma unroll
    for (int j = 0; j < 2; ++j)
      acc[i][j] = v8f{};

  // global fp32 -> LDS bf16 tile stage (128x32 each for A and B)
  auto load_tiles = [&](int buf, int k0) {
    #pragma unroll
    for (int i = 0; i < 4; ++i) {
      int id = tid + i * 256;       // 0..1023
      int r  = id >> 3;             // row 0..127
      int c  = (id & 7) << 2;       // col 0,4,...,28
      float4 fa = *(const float4*)(X + (size_t)(m0 + r) * HH + k0 + c);
      __bf16* da = &As[buf][r][c];
      da[0] = (__bf16)fa.x; da[1] = (__bf16)fa.y;
      da[2] = (__bf16)fa.z; da[3] = (__bf16)fa.w;
      float4 fb = *(const float4*)(W + (size_t)(n0 + r) * HH + k0 + c);
      __bf16* db = &Bs[buf][r][c];
      db[0] = (__bf16)fb.x; db[1] = (__bf16)fb.y;
      db[2] = (__bf16)fb.z; db[3] = (__bf16)fb.w;
    }
  };

  load_tiles(0, 0);
  __syncthreads();

  const int KT = HH / TILE_K;     // 32 k-chunks
  for (int kt = 0; kt < KT; ++kt) {
    const int cur = kt & 1;
    if (kt + 1 < KT) load_tiles(cur ^ 1, (kt + 1) * TILE_K);

    // prefetch k+2 chunk into caches (global_prefetch_b8)
    if (kt + 2 < KT) {
      const int pk = (kt + 2) * TILE_K;
      __builtin_prefetch(X + (size_t)(m0 + (tid & 127)) * HH + pk, 0, 3);
      __builtin_prefetch(W + (size_t)(n0 + (tid & 127)) * HH + pk, 0, 3);
    }

    // Fragment loads per ISA 16-bit A layout:
    //   lanes 0-15 : K 0-7  (bytes 0..15)  and K 16-23 (bytes 32..47)
    //   lanes 16-31: K 8-15 (bytes 16..31) and K 24-31 (bytes 48..63)
    v16bf afrag[4], bfrag[2];
    #pragma unroll
    for (int mi = 0; mi < 4; ++mi) {
      const __bf16* src = &As[cur][waveM * 64 + mi * 16 + r16][0];
      union { uint4 u[2]; v16bf v; } t;
      t.u[0] = *(const uint4*)(src + half * 8);
      t.u[1] = *(const uint4*)(src + 16 + half * 8);
      afrag[mi] = t.v;
    }
    #pragma unroll
    for (int ni = 0; ni < 2; ++ni) {
      const __bf16* src = &Bs[cur][waveN * 32 + ni * 16 + r16][0];
      union { uint4 u[2]; v16bf v; } t;
      t.u[0] = *(const uint4*)(src + half * 8);
      t.u[1] = *(const uint4*)(src + 16 + half * 8);
      bfrag[ni] = t.v;
    }

    #pragma unroll
    for (int mi = 0; mi < 4; ++mi)
      #pragma unroll
      for (int ni = 0; ni < 2; ++ni)
        acc[mi][ni] = __builtin_amdgcn_wmma_f32_16x16x32_bf16(
            false, afrag[mi], false, bfrag[ni],
            (short)0, acc[mi][ni], false, false);

    __syncthreads();
  }

  // Epilogue: C/D layout -> lanes 0-15: M=j, lanes 16-31: M=8+j; N=r16
  #pragma unroll
  for (int mi = 0; mi < 4; ++mi) {
    #pragma unroll
    for (int ni = 0; ni < 2; ++ni) {
      const int col = n0 + waveN * 32 + ni * 16 + r16;
      const float bv = bias[col];
      #pragma unroll
      for (int j = 0; j < 8; ++j) {
        const int row = m0 + waveM * 64 + mi * 16 + half * 8 + j;
        Out[(size_t)row * HH + col] = acc[mi][ni][j] + bv;
      }
    }
  }
}

// ---------------------------------------------------------------------------
// Global attention: q at position 0 of each (b,h) attends over all S keys.
// grid: B*NH blocks, 256 threads.
// ---------------------------------------------------------------------------
__global__ __launch_bounds__(256)
void global_attn(const float* __restrict__ Qf, const float* __restrict__ Kf,
                 const float* __restrict__ Vf, const float* __restrict__ mask,
                 float* __restrict__ ctx)
{
  const int bh = blockIdx.x;
  const int b = bh / NHH;
  const int h = bh % NHH;
  const int tid = threadIdx.x;

  __shared__ float q0[DD];
  __shared__ float sc[SS];
  __shared__ float red[256];

  const float* qptr = Qf + (size_t)(b * SS) * HH + h * DD;
  if (tid < DD) q0[tid] = qptr[tid];
  __syncthreads();

  float lmax = -INFINITY;
  for (int s = tid; s < SS; s += 256) {
    const float* kp = Kf + (size_t)(b * SS + s) * HH + h * DD;
    float d = 0.f;
    #pragma unroll 8
    for (int i = 0; i < DD; ++i) d += q0[i] * kp[i];
    d = d * ATT_SCALE + mask[b * SS + s];
    sc[s] = d;
    lmax = fmaxf(lmax, d);
  }
  red[tid] = lmax; __syncthreads();
  for (int off = 128; off > 0; off >>= 1) {
    if (tid < off) red[tid] = fmaxf(red[tid], red[tid + off]);
    __syncthreads();
  }
  const float mx = red[0];
  __syncthreads();

  float lsum = 0.f;
  for (int s = tid; s < SS; s += 256) {
    float e = __expf(sc[s] - mx);
    sc[s] = e;
    lsum += e;
  }
  red[tid] = lsum; __syncthreads();
  for (int off = 128; off > 0; off >>= 1) {
    if (tid < off) red[tid] += red[tid + off];
    __syncthreads();
  }
  const float inv = 1.f / red[0];
  __syncthreads();

  // out[d] = sum_s p[s] * V[s][d]; 4 key-groups x 64 dims
  const int d = tid & 63;
  const int g = tid >> 6;
  float oacc = 0.f;
  for (int s = g; s < SS; s += 4)
    oacc += sc[s] * Vf[(size_t)(b * SS + s) * HH + h * DD + d];
  red[tid] = oacc; __syncthreads();
  if (g == 0) {
    float o = (red[d] + red[64 + d] + red[128 + d] + red[192 + d]) * inv;
    ctx[(size_t)(b * SS + 0) * HH + h * DD + d] = o;
  }
}

// ---------------------------------------------------------------------------
// Local attention: each (b,h,s>=1) softmaxes over {self, key0}. One wave32
// per (b,h,s); lanes cover the 64-dim head.
// ---------------------------------------------------------------------------
__global__ __launch_bounds__(256)
void local_attn(const float* __restrict__ Qf, const float* __restrict__ Kf,
                const float* __restrict__ Vf, float* __restrict__ ctx,
                float* __restrict__ lattn, int nwork)
{
  const int idx  = blockIdx.x * 8 + (threadIdx.x >> 5);
  const int lane = threadIdx.x & 31;
  if (idx >= nwork) return;

  const int s  = idx % (SS - 1) + 1;
  const int bh = idx / (SS - 1);
  const int b  = bh / NHH;
  const int h  = bh % NHH;

  const float* q  = Qf + (size_t)(b * SS + s) * HH + h * DD;
  const float* ks = Kf + (size_t)(b * SS + s) * HH + h * DD;
  const float* k0 = Kf + (size_t)(b * SS) * HH + h * DD;

  const float qa = q[lane], qb = q[lane + 32];
  float dself = qa * ks[lane] + qb * ks[lane + 32];
  float dglob = qa * k0[lane] + qb * k0[lane + 32];
  #pragma unroll
  for (int off = 16; off > 0; off >>= 1) {
    dself += __shfl_xor(dself, off, 32);
    dglob += __shfl_xor(dglob, off, 32);
  }
  const float a = dself * ATT_SCALE;
  const float c = dglob * ATT_SCALE;
  const float mx = fmaxf(a, c);
  const float e0 = __expf(a - mx);
  const float e1 = __expf(c - mx);
  const float inv = 1.f / (e0 + e1);
  const float p0 = e0 * inv, p1 = e1 * inv;

  const float* vs = Vf + (size_t)(b * SS + s) * HH + h * DD;
  const float* v0 = Vf + (size_t)(b * SS) * HH + h * DD;
  float* cp = ctx + (size_t)(b * SS + s) * HH + h * DD;
  cp[lane]      = p0 * vs[lane]      + p1 * v0[lane];
  cp[lane + 32] = p0 * vs[lane + 32] + p1 * v0[lane + 32];

  if (lane == 0) {
    float* lp = lattn + ((size_t)bh * (SS - 1) + (s - 1)) * 2;
    lp[0] = p0;
    lp[1] = p1;
  }
}

// ---------------------------------------------------------------------------
extern "C" void kernel_launch(void* const* d_in, const int* in_sizes, int n_in,
                              void* d_out, int out_size, void* d_ws, size_t ws_size,
                              hipStream_t stream)
{
  const float* hidden = (const float*)d_in[0];
  const float* mask   = (const float*)d_in[1];
  const float* Wq     = (const float*)d_in[2];
  const float* bq     = (const float*)d_in[3];
  const float* Wk     = (const float*)d_in[4];
  const float* bk     = (const float*)d_in[5];
  const float* Wv     = (const float*)d_in[6];
  const float* bv     = (const float*)d_in[7];

  float* ctx   = (float*)d_out;                       // B*S*H
  float* lattn = ctx + (size_t)BB * SS * HH;          // B*NH*(S-1)*1*2

  float* Qf = (float*)d_ws;                           // 64 MiB each
  float* Kf = Qf + (size_t)MM * HH;
  float* Vf = Kf + (size_t)MM * HH;

  dim3 g1(HH / TILE_N, MM / TILE_M, 3);
  qkv_gemm_wmma<<<g1, 256, 0, stream>>>(hidden, Wq, bq, Wk, bk, Wv, bv, Qf, Kf, Vf);

  global_attn<<<BB * NHH, 256, 0, stream>>>(Qf, Kf, Vf, mask, ctx);

  const int nwork = BB * NHH * (SS - 1);
  local_attn<<<(nwork + 7) / 8, 256, 0, stream>>>(Qf, Kf, Vf, ctx, lattn, nwork);
}
